// RankPredictor_42314017800579
// MI455X (gfx1250) — matrix operand order repaired and checked by reading, not compiled, not run on hardware
//
#include <hip/hip_runtime.h>
#include <hip/hip_bf16.h>

// ---------------------------------------------------------------------------
// GATv2 GNN forward for gfx1250 (MI455X).  wave32 everywhere.
// Dense 64x64 GEMMs use v_wmma_f32_16x16x32_f16 (f16 in, f32 accumulate),
// with weights pre-packed into WMMA B-fragment-major layout so each fragment
// is a single contiguous 32B vector load.
// Edge passes are the memory-bound hot spots (atomics + gathers).
// ---------------------------------------------------------------------------

#define NNODES 100000
#define NEDGES 1600000
#define NB     16
#define HID    64
#define EDIM   32
#define GDIM   32
#define NLAYER 4
#define LN_EPS 1e-5f
#define ENC_NEG_INF 0x007FFFFFu   // fenc(-inf)

typedef __attribute__((ext_vector_type(16))) _Float16 v16h;
typedef __attribute__((ext_vector_type(8)))  _Float16 v8h;
typedef __attribute__((ext_vector_type(8)))  float    v8f;

// ---- helpers --------------------------------------------------------------
__device__ __forceinline__ float wave_sum32(float v) {
  #pragma unroll
  for (int m = 16; m > 0; m >>= 1) v += __shfl_xor(v, m, 32);
  return v;
}
__device__ __forceinline__ float lrelu(float x) { return x > 0.f ? x : 0.2f * x; }

// order-preserving float<->uint map so we can use integer atomicMax
__device__ __forceinline__ unsigned fenc(float f) {
  unsigned u = __float_as_uint(f);
  return (u & 0x80000000u) ? ~u : (u | 0x80000000u);
}
__device__ __forceinline__ float fdec(unsigned u) {
  unsigned v = (u & 0x80000000u) ? (u & 0x7FFFFFFFu) : ~u;
  return __uint_as_float(v);
}

// ---- generic fill ---------------------------------------------------------
__global__ void k_fill_u32(unsigned* p, unsigned v, long n) {
  long i = (long)blockIdx.x * blockDim.x + threadIdx.x;
  long s = (long)gridDim.x * blockDim.x;
  for (; i < n; i += s) p[i] = v;
}

// pack a 64x64 f32 weight (row-major, W[k][n]) into WMMA B-fragment-major f16:
// dst[(((ntile*2)+kt)*32 + lane)*16 + j] = W[kt*32 + (j>>3)*16 + (lane>>4)*8 + (j&7)][ntile*16 + (lane&15)]
__global__ void k_pack_b(const float* __restrict__ W, _Float16* __restrict__ dst) {
  int i = blockIdx.x * blockDim.x + threadIdx.x;  // 0..4095
  int j     = i & 15;
  int lane  = (i >> 4) & 31;
  int kt    = (i >> 9) & 1;
  int ntile = i >> 10;
  int col = ntile * 16 + (lane & 15);
  int k   = kt * 32 + (j >> 3) * 16 + (lane >> 4) * 8 + (j & 7);
  dst[i] = (_Float16)W[k * 64 + col];
}

// ---- encoders -------------------------------------------------------------
// node: [N,8] @ [8,64] + b -> LN -> relu.  wave per node, lane owns feats l,l+32
__global__ void k_node_enc(const float* __restrict__ x, const float* __restrict__ W,
                           const float* __restrict__ b, const float* __restrict__ g,
                           const float* __restrict__ be,
                           float* __restrict__ h, _Float16* __restrict__ hh) {
  long gid = (long)blockIdx.x * blockDim.x + threadIdx.x;
  int n = (int)(gid >> 5);
  int lane = threadIdx.x & 31;
  if (n >= NNODES) return;
  float xi[8];
  #pragma unroll
  for (int k = 0; k < 8; k++) xi[k] = x[n * 8 + k];
  float v0 = b[lane], v1 = b[lane + 32];
  #pragma unroll
  for (int k = 0; k < 8; k++) {
    v0 += xi[k] * W[k * 64 + lane];
    v1 += xi[k] * W[k * 64 + lane + 32];
  }
  float mean = wave_sum32(v0 + v1) * (1.f / 64.f);
  float d0 = v0 - mean, d1 = v1 - mean;
  float var = wave_sum32(d0 * d0 + d1 * d1) * (1.f / 64.f);
  float rs = rsqrtf(var + LN_EPS);
  float o0 = fmaxf(d0 * rs * g[lane] + be[lane], 0.f);
  float o1 = fmaxf(d1 * rs * g[lane + 32] + be[lane + 32], 0.f);
  h[(long)n * 64 + lane] = o0;
  h[(long)n * 64 + lane + 32] = o1;
  hh[(long)n * 64 + lane] = (_Float16)o0;
  hh[(long)n * 64 + lane + 32] = (_Float16)o1;
}

// edge: [E,2] @ [2,32] + b -> LN -> relu.  wave per edge, lane = channel
__global__ void k_edge_enc(const float* __restrict__ ea, const float* __restrict__ W,
                           const float* __restrict__ b, const float* __restrict__ g,
                           const float* __restrict__ be, float* __restrict__ e_full) {
  long gid = (long)blockIdx.x * blockDim.x + threadIdx.x;
  long i = gid >> 5;
  int lane = threadIdx.x & 31;
  if (i >= NEDGES) return;
  float a0 = ea[i * 2 + 0], a1 = ea[i * 2 + 1];
  float v = b[lane] + a0 * W[0 * 32 + lane] + a1 * W[1 * 32 + lane];
  float mean = wave_sum32(v) * (1.f / 32.f);
  float d = v - mean;
  float var = wave_sum32(d * d) * (1.f / 32.f);
  float o = fmaxf(d * rsqrtf(var + LN_EPS) * g[lane] + be[lane], 0.f);
  e_full[i * 32 + lane] = o;
}

// global: [B,5] @ [5,32] + b -> LN -> relu.  single block, wave per row
__global__ void k_glob_enc(const float* __restrict__ ga, const float* __restrict__ W,
                           const float* __restrict__ b, const float* __restrict__ g,
                           const float* __restrict__ be, float* __restrict__ gout) {
  int row = threadIdx.x >> 5;
  int lane = threadIdx.x & 31;
  if (row >= NB) return;
  float v = b[lane];
  #pragma unroll
  for (int k = 0; k < 5; k++) v += ga[row * 5 + k] * W[k * 32 + lane];
  float mean = wave_sum32(v) * (1.f / 32.f);
  float d = v - mean;
  float var = wave_sum32(d * d) * (1.f / 32.f);
  float o = fmaxf(d * rsqrtf(var + LN_EPS) * g[lane] + be[lane], 0.f);
  gout[row * 32 + lane] = o;
}

// ---- self-loop edge attr (mean of incoming encoded edges) -----------------
__global__ void k_loop_accum(float* __restrict__ e_full, const int* __restrict__ dst0,
                             float* __restrict__ cnt) {
  long gid = (long)blockIdx.x * blockDim.x + threadIdx.x;
  long i = gid >> 5;
  int lane = threadIdx.x & 31;
  if (i >= NEDGES) return;
  int d = dst0[i];
  atomicAdd(&e_full[((long)NEDGES + d) * 32 + lane], e_full[i * 32 + lane]);
  if (lane == 0) atomicAdd(&cnt[d], 1.f);
}

__global__ void k_loop_fin(float* __restrict__ e_full, const float* __restrict__ cnt) {
  long gid = (long)blockIdx.x * blockDim.x + threadIdx.x;
  int n = (int)(gid >> 5);
  int lane = threadIdx.x & 31;
  if (n >= NNODES) return;
  e_full[((long)NEDGES + n) * 32 + lane] /= fmaxf(cnt[n], 1.f);
}

// ---- WMMA GEMM: xl = h@Wl+bl, xr = h@Wr+br --------------------------------
// block = 16-row M-tile, 8 waves = {Wl,Wr} x 4 column tiles.
// A-fragment (16-bit A 16x32 ISA layout): lane m|hi -> halves 0..7: K=kb+hi*8+j,
// halves 8..15: K=kb+16+hi*8+j  (two contiguous 16B loads from hh).
// B-fragment: pre-packed fragment-major (one contiguous 32B load per frag).
__global__ void k_gemm_xlxr(const _Float16* __restrict__ hh,
                            const _Float16* __restrict__ Wl16,
                            const _Float16* __restrict__ Wr16,
                            const float* __restrict__ bl, const float* __restrict__ br,
                            float* __restrict__ xl, float* __restrict__ xr) {
  int mtile = blockIdx.x;
  int w = threadIdx.x >> 5;
  int lane = threadIdx.x & 31;
  int mat = w >> 2;        // 0 -> Wl, 1 -> Wr
  int ntile = w & 3;
  const _Float16* W = mat ? Wr16 : Wl16;
  const float* bias = mat ? br : bl;
  float* out = mat ? xr : xl;
  int row0 = mtile * 16;
  int m = row0 + (lane & 15);
  int hi = lane >> 4;
  int col = ntile * 16 + (lane & 15);

  v8f acc;
  float bv = bias[col];
  #pragma unroll
  for (int r = 0; r < 8; r++) acc[r] = bv;

  #pragma unroll
  for (int kt = 0; kt < 2; kt++) {
    int kb = kt * 32;
    const _Float16* arow = hh + (long)m * 64 + kb + hi * 8;
    v8h alo = *(const v8h*)(arow);
    v8h ahi = *(const v8h*)(arow + 16);
    v16h a = __builtin_shufflevector(alo, ahi, 0, 1, 2, 3, 4, 5, 6, 7,
                                     8, 9, 10, 11, 12, 13, 14, 15);
    v16h bf = *(const v16h*)(W + (((ntile * 2 + kt) * 32 + lane) << 4));
    acc = __builtin_amdgcn_wmma_f32_16x16x32_f16(false, a, false, bf,
                                                 (short)0, acc, false, false);
  }
  #pragma unroll
  for (int r = 0; r < 8; r++) out[(long)(row0 + hi * 8 + r) * 64 + col] = acc[r];
}

// ---- attention logits + segment max ---------------------------------------
// wave per edge (incl. self loops at i >= E).  lane owns feats l, l+32.
__global__ void k_logits(const float* __restrict__ xl, const float* __restrict__ xr,
                         const float* __restrict__ e_full, const float* __restrict__ We,
                         const float* __restrict__ att,
                         const int* __restrict__ src0, const int* __restrict__ dst0,
                         float* __restrict__ logits, unsigned* __restrict__ lmaxU) {
  long gid = (long)blockIdx.x * blockDim.x + threadIdx.x;
  long i = gid >> 5;
  int lane = threadIdx.x & 31;
  if (i >= (long)NEDGES + NNODES) return;
  int s, d;
  if (i < NEDGES) { s = src0[i]; d = dst0[i]; } else { s = d = (int)(i - NEDGES); }
  int f0 = lane, f1 = lane + 32;
  float a0 = xl[(long)s * 64 + f0] + xr[(long)d * 64 + f0];
  float a1 = xl[(long)s * 64 + f1] + xr[(long)d * 64 + f1];
  const float* er = e_full + i * 32;
  #pragma unroll
  for (int c = 0; c < 32; c++) {
    float ev = er[c];
    a0 += ev * We[c * 64 + f0];
    a1 += ev * We[c * 64 + f1];
  }
  a0 = lrelu(a0);
  a1 = lrelu(a1);
  int c16 = lane & 15, hi = lane >> 4;
  float p0 = a0 * att[hi * 16 + c16];         // heads 0/1
  float p1 = a1 * att[(2 + hi) * 16 + c16];   // heads 2/3
  #pragma unroll
  for (int mm = 8; mm > 0; mm >>= 1) {
    p0 += __shfl_xor(p0, mm, 32);
    p1 += __shfl_xor(p1, mm, 32);
  }
  if (c16 == 0) {
    logits[i * 4 + hi] = p0;
    logits[i * 4 + 2 + hi] = p1;
    atomicMax(&lmaxU[(long)d * 4 + hi], fenc(p0));
    atomicMax(&lmaxU[(long)d * 4 + 2 + hi], fenc(p1));
  }
}

// ---- softmax numerator scatter + denominator ------------------------------
__global__ void k_scatter(const float* __restrict__ xl, const float* __restrict__ logits,
                          const unsigned* __restrict__ lmaxU, float* __restrict__ den,
                          float* __restrict__ out_raw,
                          const int* __restrict__ src0, const int* __restrict__ dst0) {
  long gid = (long)blockIdx.x * blockDim.x + threadIdx.x;
  long i = gid >> 5;
  int lane = threadIdx.x & 31;
  if (i >= (long)NEDGES + NNODES) return;
  int s, d;
  if (i < NEDGES) { s = src0[i]; d = dst0[i]; } else { s = d = (int)(i - NEDGES); }
  int hi = lane >> 4;
  int h0 = hi, h1 = 2 + hi;
  float ex0 = __expf(logits[i * 4 + h0] - fdec(lmaxU[(long)d * 4 + h0]));
  float ex1 = __expf(logits[i * 4 + h1] - fdec(lmaxU[(long)d * 4 + h1]));
  if ((lane & 15) == 0) {
    atomicAdd(&den[(long)d * 4 + h0], ex0);
    atomicAdd(&den[(long)d * 4 + h1], ex1);
  }
  atomicAdd(&out_raw[(long)d * 64 + lane],      ex0 * xl[(long)s * 64 + lane]);
  atomicAdd(&out_raw[(long)d * 64 + lane + 32], ex1 * xl[(long)s * 64 + lane + 32]);
}

// ---- per-node finalize: normalize + bias + LN + leaky + residual ----------
__global__ void k_finalize(const float* __restrict__ out_raw, const float* __restrict__ den,
                           const float* __restrict__ bias, const float* __restrict__ g,
                           const float* __restrict__ be, const float* __restrict__ h_res,
                           float* __restrict__ h_out, _Float16* __restrict__ hh) {
  long gid = (long)blockIdx.x * blockDim.x + threadIdx.x;
  int n = (int)(gid >> 5);
  int lane = threadIdx.x & 31;
  if (n >= NNODES) return;
  int hi = lane >> 4;  // head of feature `lane` (0..31 -> head 0/1)
  float v0 = out_raw[(long)n * 64 + lane]      / den[(long)n * 4 + hi]     + bias[lane];
  float v1 = out_raw[(long)n * 64 + lane + 32] / den[(long)n * 4 + 2 + hi] + bias[lane + 32];
  float mean = wave_sum32(v0 + v1) * (1.f / 64.f);
  float d0 = v0 - mean, d1 = v1 - mean;
  float var = wave_sum32(d0 * d0 + d1 * d1) * (1.f / 64.f);
  float rs = rsqrtf(var + LN_EPS);
  float o0 = lrelu(d0 * rs * g[lane] + be[lane])           + h_res[(long)n * 64 + lane];
  float o1 = lrelu(d1 * rs * g[lane + 32] + be[lane + 32]) + h_res[(long)n * 64 + lane + 32];
  h_out[(long)n * 64 + lane] = o0;
  h_out[(long)n * 64 + lane + 32] = o1;
  hh[(long)n * 64 + lane] = (_Float16)o0;
  hh[(long)n * 64 + lane + 32] = (_Float16)o1;
}

// ---- batch pooling --------------------------------------------------------
__global__ void k_pool(const float* __restrict__ h, const int* __restrict__ batch,
                       float* __restrict__ psum, unsigned* __restrict__ pmaxU,
                       float* __restrict__ bc) {
  long gid = (long)blockIdx.x * blockDim.x + threadIdx.x;
  int n = (int)(gid >> 5);
  int lane = threadIdx.x & 31;
  if (n >= NNODES) return;
  int b = batch[n];
  float v0 = h[(long)n * 64 + lane], v1 = h[(long)n * 64 + lane + 32];
  atomicAdd(&psum[b * 64 + lane], v0);
  atomicAdd(&psum[b * 64 + lane + 32], v1);
  atomicMax(&pmaxU[b * 64 + lane], fenc(v0));
  atomicMax(&pmaxU[b * 64 + lane + 32], fenc(v1));
  if (lane == 0) atomicAdd(&bc[b], 1.f);
}

// ---- head MLP: [mean|max|g](160) @ W1(160,64) -> relu -> @ W2(64,1) -> exp
__global__ void k_head(const float* __restrict__ psum, const unsigned* __restrict__ pmaxU,
                       const float* __restrict__ bc, const float* __restrict__ gbuf,
                       const float* __restrict__ W1, const float* __restrict__ b1,
                       const float* __restrict__ W2, const float* __restrict__ b2,
                       float* __restrict__ out) {
  int b = threadIdx.x >> 5;
  int lane = threadIdx.x & 31;
  if (b >= NB) return;
  float inv = 1.f / fmaxf(bc[b], 1.f);
  float h0 = b1[lane], h1 = b1[lane + 32];
  #pragma unroll 4
  for (int k = 0; k < 64; k++) {
    float f = psum[b * 64 + k] * inv;
    h0 += f * W1[k * 64 + lane];
    h1 += f * W1[k * 64 + lane + 32];
  }
  #pragma unroll 4
  for (int k = 0; k < 64; k++) {
    float f = fdec(pmaxU[b * 64 + k]);
    h0 += f * W1[(64 + k) * 64 + lane];
    h1 += f * W1[(64 + k) * 64 + lane + 32];
  }
  #pragma unroll 4
  for (int k = 0; k < 32; k++) {
    float f = gbuf[b * 32 + k];
    h0 += f * W1[(128 + k) * 64 + lane];
    h1 += f * W1[(128 + k) * 64 + lane + 32];
  }
  h0 = fmaxf(h0, 0.f);
  h1 = fmaxf(h1, 0.f);
  float p = h0 * W2[lane] + h1 * W2[lane + 32];
  p = wave_sum32(p);
  if (lane == 0) out[b] = __expf(p + b2[0]);
}

// ---------------------------------------------------------------------------
extern "C" void kernel_launch(void* const* d_in, const int* in_sizes, int n_in,
                              void* d_out, int out_size, void* d_ws, size_t ws_size,
                              hipStream_t stream) {
  (void)in_sizes; (void)out_size; (void)ws_size;
  const float* x     = (const float*)d_in[0];
  const int*   ei    = (const int*)d_in[1];
  const float* ea    = (const float*)d_in[2];
  const int*   batch = (const int*)d_in[3];
  const float* ga    = (const float*)d_in[4];
  const int* src0 = ei;
  const int* dst0 = ei + NEDGES;

  // -- param leaves in jax sorted-key flatten order ------------------------
  static const int leaf_sizes[52] = {
      32, 32, 64, 32,                    // edge_enc: be, g, lin.W, lin.b
      32, 32, 160, 32,                   // glob_enc: be, g, lin.W, lin.b
      10240, 64, 64, 1,                  // head: lin1.W, lin1.b, lin2.W, lin2.b
      2048, 4096, 4096, 64, 64, 64, 64, 64, 64,  // layer0: We,Wl,Wr,att,be,bias,bl,br,g
      2048, 4096, 4096, 64, 64, 64, 64, 64, 64,  // layer1
      2048, 4096, 4096, 64, 64, 64, 64, 64, 64,  // layer2
      2048, 4096, 4096, 64, 64, 64, 64, 64, 64,  // layer3
      64, 64, 512, 64};                  // node_enc: be, g, lin.W, lin.b
  const float* P[52];
  if (n_in >= 5 + 52) {
    for (int i = 0; i < 52; i++) P[i] = (const float*)d_in[5 + i];
  } else {
    const float* base = (const float*)d_in[5];
    long off = 0;
    for (int i = 0; i < 52; i++) { P[i] = base + off; off += leaf_sizes[i]; }
  }
  const float* ee_be = P[0];  const float* ee_g = P[1];
  const float* ee_W  = P[2];  const float* ee_b = P[3];
  const float* ge_be = P[4];  const float* ge_g = P[5];
  const float* ge_W  = P[6];  const float* ge_b = P[7];
  const float* hW1 = P[8]; const float* hb1 = P[9];
  const float* hW2 = P[10]; const float* hb2 = P[11];
  const float* ne_be = P[48]; const float* ne_g = P[49];
  const float* ne_W  = P[50]; const float* ne_b = P[51];

  // -- workspace carve-up --------------------------------------------------
  char* wp = (char*)d_ws;
  auto alloc = [&](size_t bytes) -> void* {
    void* p = (void*)wp;
    wp += (bytes + 255) & ~(size_t)255;
    return p;
  };
  float*     h0      = (float*)alloc((size_t)NNODES * 64 * 4);
  float*     h1      = (float*)alloc((size_t)NNODES * 64 * 4);
  _Float16*  hh      = (_Float16*)alloc((size_t)NNODES * 64 * 2);
  float*     xl      = (float*)alloc((size_t)NNODES * 64 * 4);
  float*     xr      = (float*)alloc((size_t)NNODES * 64 * 4);
  float*     out_raw = (float*)alloc((size_t)NNODES * 64 * 4);
  float*     e_full  = (float*)alloc((size_t)(NEDGES + NNODES) * 32 * 4);
  float*     logits  = (float*)alloc((size_t)(NEDGES + NNODES) * 4 * 4);
  unsigned*  lmaxU   = (unsigned*)alloc((size_t)NNODES * 4 * 4);
  float*     den     = (float*)alloc((size_t)NNODES * 4 * 4);
  float*     cnt     = (float*)alloc((size_t)NNODES * 4);
  _Float16*  w16     = (_Float16*)alloc((size_t)NLAYER * 2 * 4096 * 2);
  float*     psum    = (float*)alloc((size_t)NB * 64 * 4);
  unsigned*  pmaxU   = (unsigned*)alloc((size_t)NB * 64 * 4);
  float*     bc      = (float*)alloc((size_t)NB * 4);
  float*     gbuf    = (float*)alloc((size_t)NB * 32 * 4);

  const int TB = 256;
  const int NODE_BLK = NNODES / 8;            // 12500 (wave per node)
  const int EDGE_BLK = NEDGES / 8;            // 200000 (wave per edge)
  const int FULL_BLK = (NEDGES + NNODES) / 8; // 212500

  // pre-pack Wl/Wr of each layer into WMMA B-fragment-major f16
  for (int l = 0; l < NLAYER; l++) {
    const float* Wl = P[12 + l * 9 + 1];
    const float* Wr = P[12 + l * 9 + 2];
    k_pack_b<<<16, TB, 0, stream>>>(Wl, w16 + (size_t)l * 8192);
    k_pack_b<<<16, TB, 0, stream>>>(Wr, w16 + (size_t)l * 8192 + 4096);
  }

  // encoders
  k_node_enc<<<NODE_BLK, TB, 0, stream>>>(x, ne_W, ne_b, ne_g, ne_be, h0, hh);
  k_edge_enc<<<EDGE_BLK, TB, 0, stream>>>(ea, ee_W, ee_b, ee_g, ee_be, e_full);
  k_glob_enc<<<1, 512, 0, stream>>>(ga, ge_W, ge_b, ge_g, ge_be, gbuf);

  // self-loop edge attrs (mean of incoming)
  k_fill_u32<<<2048, TB, 0, stream>>>((unsigned*)cnt, 0u, NNODES);
  k_fill_u32<<<4096, TB, 0, stream>>>((unsigned*)(e_full + (size_t)NEDGES * 32), 0u,
                                      (long)NNODES * 32);
  k_loop_accum<<<EDGE_BLK, TB, 0, stream>>>(e_full, dst0, cnt);
  k_loop_fin<<<NODE_BLK, TB, 0, stream>>>(e_full, cnt);

  // GATv2 layers
  float* h_cur = h0;
  float* h_nxt = h1;
  for (int l = 0; l < NLAYER; l++) {
    const float* We   = P[12 + l * 9 + 0];
    const float* att  = P[12 + l * 9 + 3];
    const float* lbe  = P[12 + l * 9 + 4];
    const float* bias = P[12 + l * 9 + 5];
    const float* bl   = P[12 + l * 9 + 6];
    const float* br   = P[12 + l * 9 + 7];
    const float* lg   = P[12 + l * 9 + 8];

    k_gemm_xlxr<<<NNODES / 16, TB, 0, stream>>>(hh, w16 + (size_t)l * 8192,
                                                w16 + (size_t)l * 8192 + 4096,
                                                bl, br, xl, xr);
    k_fill_u32<<<4096, TB, 0, stream>>>((unsigned*)out_raw, 0u, (long)NNODES * 64);
    k_fill_u32<<<2048, TB, 0, stream>>>((unsigned*)den, 0u, (long)NNODES * 4);
    k_fill_u32<<<2048, TB, 0, stream>>>(lmaxU, ENC_NEG_INF, (long)NNODES * 4);

    k_logits<<<FULL_BLK, TB, 0, stream>>>(xl, xr, e_full, We, att, src0, dst0,
                                          logits, lmaxU);
    k_scatter<<<FULL_BLK, TB, 0, stream>>>(xl, logits, lmaxU, den, out_raw,
                                           src0, dst0);
    k_finalize<<<NODE_BLK, TB, 0, stream>>>(out_raw, den, bias, lg, lbe,
                                            h_cur, h_nxt, hh);
    float* t = h_cur; h_cur = h_nxt; h_nxt = t;
  }

  // pooling + head
  k_fill_u32<<<16, TB, 0, stream>>>((unsigned*)psum, 0u, (long)NB * 64);
  k_fill_u32<<<16, TB, 0, stream>>>(pmaxU, ENC_NEG_INF, (long)NB * 64);
  k_fill_u32<<<1, TB, 0, stream>>>((unsigned*)bc, 0u, NB);
  k_pool<<<NODE_BLK, TB, 0, stream>>>(h_cur, batch, psum, pmaxU, bc);
  k_head<<<1, 512, 0, stream>>>(psum, pmaxU, bc, gbuf, hW1, hb1, hW2, hb2,
                                (float*)d_out);
}
